// MambaMixer_9423158247761
// MI455X (gfx1250) — matrix operand order repaired and checked by reading, not compiled, not run on hardware
//
#include <hip/hip_runtime.h>
#include <hip/hip_bf16.h>
#include <math.h>

// ---- problem constants (match reference) ----
#define BATCH 2
#define SEQ   2048
#define HID   2048
#define INTER 4096
#define STATE 16
#define RANK  128
#define KCONV 4
#define ML    (BATCH * SEQ)          // 4096 rows of activations
#define PROJW (2 * INTER)            // 8192
#define XPN   (RANK + 2 * STATE)     // 160

typedef __bf16        v16bf __attribute__((ext_vector_type(16)));
typedef float         v8f   __attribute__((ext_vector_type(8)));
typedef unsigned int  v8u   __attribute__((ext_vector_type(8)));

// Build a 16-element bf16 fragment from two 16-byte (b128) loads. Pure SSA —
// no unions, no stack traffic.
__device__ __forceinline__ v16bf make_frag(uint4 a, uint4 b) {
    v8u w;
    w[0] = a.x; w[1] = a.y; w[2] = a.z; w[3] = a.w;
    w[4] = b.x; w[5] = b.y; w[6] = b.z; w[7] = b.w;
    return __builtin_bit_cast(v16bf, w);
}

// fp32 -> bf16 hi/lo planes (error-compensated split), memory-bound prepass.
__global__ void split_kernel(const float* __restrict__ x,
                             __bf16* __restrict__ h,
                             __bf16* __restrict__ l, long long n) {
    const long long i = (long long)blockIdx.x * blockDim.x + threadIdx.x;
    if (i >= n) return;
    const float v = x[i];
    const __bf16 hv = (__bf16)v;
    h[i] = hv;
    l[i] = (__bf16)(v - (float)hv);
}

// C = A(MxK, lda) * W^T (W is NxK row-major, ldw), fp32 accumulate, with
// pre-split bf16 hi/lo planes: product ~= Ah*Bh + Al*Bh + Ah*Bl (rel err ~2^-16).
// Wave computes a 16(M) x (JT*16)(N) tile; JT is compile-time (2 or 4).
// EPI: 0 = plain store, 1 = softplus(C + bias[n]).
template <int EPI, int JT>
__global__ void wmma_gemm_kernel(const __bf16* __restrict__ AH,
                                 const __bf16* __restrict__ AL,
                                 const __bf16* __restrict__ WH,
                                 const __bf16* __restrict__ WL,
                                 float* __restrict__ C,
                                 const float* __restrict__ bias,
                                 int M, int K, int lda, int ldw, int ldc, int nOff) {
    const int lane   = threadIdx.x;          // 0..31 (wave32)
    const int laneN  = lane & 15;
    const int laneHi = lane >> 4;
    const int mBase  = (blockIdx.y * blockDim.y + threadIdx.y) * 16;  // wave-uniform
    const int nBase  = nOff + blockIdx.x * (JT * 16);
    if (mBase >= M) return;                   // wave-uniform guard (grids are exact)

    v8f acc[JT];
#pragma unroll
    for (int j = 0; j < JT; ++j) { v8f z = {}; acc[j] = z; }

    // A fragment: lane = M row (lane&15). bf16 K-pairs are dword-packed, so the
    // per-lane pattern is 8 consecutive dwords (two b128 loads).
    const unsigned int* ArowH = (const unsigned int*)(AH + (size_t)(mBase + laneN) * lda);
    const unsigned int* ArowL = (const unsigned int*)(AL + (size_t)(mBase + laneN) * lda);
    // B fragment: lane = N col (lane&15); 8 consecutive dwords (two b128 loads).
    const unsigned int* WrowH = (const unsigned int*)(WH + (size_t)(nBase + laneN) * ldw);
    const unsigned int* WrowL = (const unsigned int*)(WL + (size_t)(nBase + laneN) * ldw);
    const size_t wsub = (size_t)8 * ldw;      // dword stride between 16-row N-subtiles

    const int aOff = laneHi * 8;              // (laneHi*16)/2 dwords
    const int bOff = laneHi * 8;              // (laneHi*16)/2 dwords
    const int Kd   = K >> 1;                  // K in dwords

    for (int kt2 = 0; kt2 < Kd; kt2 += 16) {
        const v16bf ah = make_frag(*(const uint4*)(ArowH + kt2 + aOff),
                                   *(const uint4*)(ArowH + kt2 + 4 + aOff));
        const v16bf al = make_frag(*(const uint4*)(ArowL + kt2 + aOff),
                                   *(const uint4*)(ArowL + kt2 + 4 + aOff));
        if (kt2 + 16 < Kd) __builtin_prefetch(ArowH + kt2 + 16, 0, 1);
#pragma unroll
        for (int jp = 0; jp < JT; jp += 2) {
            const unsigned int* wh0 = WrowH + (size_t)jp * wsub + kt2 + bOff;
            const unsigned int* wh1 = wh0 + wsub;
            const unsigned int* wl0 = WrowL + (size_t)jp * wsub + kt2 + bOff;
            const unsigned int* wl1 = wl0 + wsub;
            const v16bf bh0 = make_frag(*(const uint4*)(wh0), *(const uint4*)(wh0 + 4));
            const v16bf bh1 = make_frag(*(const uint4*)(wh1), *(const uint4*)(wh1 + 4));
            const v16bf bl0 = make_frag(*(const uint4*)(wl0), *(const uint4*)(wl0 + 4));
            const v16bf bl1 = make_frag(*(const uint4*)(wl1), *(const uint4*)(wl1 + 4));
            // Alternate accumulators between consecutive WMMAs: no D-operand RAW
            // between adjacent matrix ops.
            acc[jp]     = __builtin_amdgcn_wmma_f32_16x16x32_bf16(
                false, ah, false, bh0, (short)0, acc[jp],     false, false);
            acc[jp + 1] = __builtin_amdgcn_wmma_f32_16x16x32_bf16(
                false, ah, false, bh1, (short)0, acc[jp + 1], false, false);
            acc[jp]     = __builtin_amdgcn_wmma_f32_16x16x32_bf16(
                false, al, false, bh0, (short)0, acc[jp],     false, false);
            acc[jp + 1] = __builtin_amdgcn_wmma_f32_16x16x32_bf16(
                false, al, false, bh1, (short)0, acc[jp + 1], false, false);
            acc[jp]     = __builtin_amdgcn_wmma_f32_16x16x32_bf16(
                false, ah, false, bl0, (short)0, acc[jp],     false, false);
            acc[jp + 1] = __builtin_amdgcn_wmma_f32_16x16x32_bf16(
                false, ah, false, bl1, (short)0, acc[jp + 1], false, false);
        }
    }

    // C/D layout: lane holds col n = laneN; VGPR r holds row r + 8*laneHi.
#pragma unroll
    for (int j = 0; j < JT; ++j) {
        const int n = nBase + j * 16 + laneN;
#pragma unroll
        for (int r = 0; r < 8; ++r) {
            const int m = mBase + r + 8 * laneHi;
            float v = acc[j][r];
            if (EPI == 1) {
                v += bias[n];
                v = (v > 20.f) ? v : log1pf(expf(v));   // softplus
            }
            C[(size_t)m * ldc + n] = v;
        }
    }
}

// Depthwise causal conv1d (K=4) + SiLU. Reads u half of proj (B,L,2*INTER),
// writes xconv (B,L,INTER).
__global__ void conv_silu_kernel(const float* __restrict__ proj,
                                 const float* __restrict__ cw,
                                 const float* __restrict__ cb,
                                 float* __restrict__ xconv) {
    const size_t idx = (size_t)blockIdx.x * blockDim.x + threadIdx.x;
    const size_t total = (size_t)ML * INTER;
    if (idx >= total) return;
    const int d = (int)(idx & (INTER - 1));
    const size_t bl = idx >> 12;             // b*SEQ + l   (INTER = 4096 = 2^12)
    const int l = (int)(bl & (SEQ - 1));     // SEQ = 2048 = 2^11
    float acc = cb[d];
#pragma unroll
    for (int k = 0; k < KCONV; ++k) {
        const int l2 = l + k - (KCONV - 1);
        if (l2 >= 0)
            acc += proj[(bl + (size_t)(k - (KCONV - 1))) * PROJW + d] * cw[d * KCONV + k];
    }
    xconv[idx] = acc / (1.f + expf(-acc));   // silu
}

// Selective scan: one thread per (b, d) channel, state[16] in registers.
// B/C row staged through LDS per timestep. Fuses skip (u*D) and gating
// (silu of gate half of proj). Writes in place over dt buffer (yg).
__global__ void scan_kernel(const float* __restrict__ proj,
                            const float* __restrict__ xconv,
                            const float* __restrict__ ssm_p,
                            const float* __restrict__ A_log,
                            const float* __restrict__ Dvec,
                            float* __restrict__ yg /* in: dt, out: gated y */) {
    const int b = blockIdx.y;
    const int d = blockIdx.x * blockDim.x + threadIdx.x;   // 0..INTER-1
    __shared__ float sB[STATE];
    __shared__ float sC[STATE];

    float a[STATE], st[STATE];
#pragma unroll
    for (int n = 0; n < STATE; ++n) {
        a[n]  = -expf(A_log[d * STATE + n]);
        st[n] = 0.f;
    }
    const float Dd = Dvec[d];

    for (int l = 0; l < SEQ; ++l) {
        const size_t row = (size_t)b * SEQ + l;
        if (threadIdx.x < 2 * STATE) {
            const float v = ssm_p[row * XPN + RANK + threadIdx.x];
            if (threadIdx.x < STATE) sB[threadIdx.x] = v;
            else                     sC[threadIdx.x - STATE] = v;
        }
        __syncthreads();

        const float dtd = yg[row * INTER + d];       // softplus(dt) from GEMM
        const float ud  = xconv[row * INTER + d];
        const float du  = dtd * ud;
        float y = 0.f;
#pragma unroll
        for (int n = 0; n < STATE; ++n) {
            st[n] = expf(dtd * a[n]) * st[n] + du * sB[n];
            y += st[n] * sC[n];
        }
        const float g  = proj[row * PROJW + INTER + d];
        const float yv = y + ud * Dd;
        yg[row * INTER + d] = yv * (g / (1.f + expf(-g)));   // overwrite dt slot
        __syncthreads();
    }
}

static inline unsigned nblk(long long n, int t) { return (unsigned)((n + t - 1) / t); }

extern "C" void kernel_launch(void* const* d_in, const int* in_sizes, int n_in,
                              void* d_out, int out_size, void* d_ws, size_t ws_size,
                              hipStream_t stream) {
    const float* hs        = (const float*)d_in[0];  // (B,L,HID)
    const float* in_proj_w = (const float*)d_in[1];  // (2*INTER, HID)
    const float* conv_w    = (const float*)d_in[2];  // (INTER,1,K)
    const float* conv_b    = (const float*)d_in[3];  // (INTER)
    const float* x_proj_w  = (const float*)d_in[4];  // (160, INTER)
    const float* dt_proj_w = (const float*)d_in[5];  // (INTER, RANK)
    const float* dt_proj_b = (const float*)d_in[6];  // (INTER)
    const float* A_log     = (const float*)d_in[7];  // (INTER, STATE)
    const float* Dvec      = (const float*)d_in[8];  // (INTER)
    const float* out_proj_w= (const float*)d_in[9];  // (HID, INTER)
    float* out = (float*)d_out;

    // ---- workspace layout ----
    float* ws    = (float*)d_ws;
    float* proj  = ws;                                   // ML * 8192 f32
    float* xconv = proj  + (size_t)ML * PROJW;           // ML * 4096 f32
    float* ssm_p = xconv + (size_t)ML * INTER;           // ML * 160  f32
    float* dtbuf = ssm_p + (size_t)ML * XPN;             // ML * 4096 f32 (dt -> gated y)
    // bf16 hi/lo arena (reused per GEMM). Slot = largest operand (16.8M elems).
    const size_t SLOT = (size_t)PROJW * HID;             // 16,777,216
    __bf16* AHp = (__bf16*)(dtbuf + (size_t)ML * INTER);
    __bf16* ALp = AHp + SLOT;
    __bf16* WHp = ALp + SLOT;
    __bf16* WLp = WHp + SLOT;

    const dim3 blk(32, 4);                               // 4 waves, 1 M-tile each

    // 1) in_proj: proj = hs @ in_proj_w^T     (4096 x 8192 x 2048)
    split_kernel<<<nblk((long long)ML * HID, 256), 256, 0, stream>>>(
        hs, AHp, ALp, (long long)ML * HID);
    split_kernel<<<nblk((long long)PROJW * HID, 256), 256, 0, stream>>>(
        in_proj_w, WHp, WLp, (long long)PROJW * HID);
    wmma_gemm_kernel<0, 4><<<dim3(PROJW / 64, ML / 64), blk, 0, stream>>>(
        AHp, ALp, WHp, WLp, proj, nullptr, ML, HID, HID, HID, PROJW, 0);

    // 2) depthwise causal conv + silu -> xconv
    conv_silu_kernel<<<nblk((long long)ML * INTER, 256), 256, 0, stream>>>(
        proj, conv_w, conv_b, xconv);

    // 3) x_proj: ssm_p = xconv @ x_proj_w^T   (4096 x 160 x 4096)
    split_kernel<<<nblk((long long)ML * INTER, 256), 256, 0, stream>>>(
        xconv, AHp, ALp, (long long)ML * INTER);
    split_kernel<<<nblk((long long)XPN * INTER, 256), 256, 0, stream>>>(
        x_proj_w, WHp, WLp, (long long)XPN * INTER);
    wmma_gemm_kernel<0, 4><<<dim3(2, ML / 64), blk, 0, stream>>>(       // n = 0..127
        AHp, ALp, WHp, WLp, ssm_p, nullptr, ML, INTER, INTER, INTER, XPN, 0);
    wmma_gemm_kernel<0, 2><<<dim3(1, ML / 64), blk, 0, stream>>>(       // n = 128..159
        AHp, ALp, WHp, WLp, ssm_p, nullptr, ML, INTER, INTER, INTER, XPN, 128);

    // 4) dt_proj + bias + softplus: dt = softplus(ssm_p[:, :128] @ dt_proj_w^T + b)
    split_kernel<<<nblk((long long)ML * XPN, 256), 256, 0, stream>>>(
        ssm_p, AHp, ALp, (long long)ML * XPN);
    split_kernel<<<nblk((long long)INTER * RANK, 256), 256, 0, stream>>>(
        dt_proj_w, WHp, WLp, (long long)INTER * RANK);
    wmma_gemm_kernel<1, 4><<<dim3(INTER / 64, ML / 64), blk, 0, stream>>>(
        AHp, ALp, WHp, WLp, dtbuf, dt_proj_b, ML, RANK, XPN, RANK, INTER, 0);

    // 5) selective scan + skip + gating (in place over dtbuf)
    scan_kernel<<<dim3(INTER / 256, BATCH), 256, 0, stream>>>(
        proj, xconv, ssm_p, A_log, Dvec, dtbuf);

    // 6) out_proj: out = yg @ out_proj_w^T    (4096 x 2048 x 4096)
    split_kernel<<<nblk((long long)ML * INTER, 256), 256, 0, stream>>>(
        dtbuf, AHp, ALp, (long long)ML * INTER);
    split_kernel<<<nblk((long long)HID * INTER, 256), 256, 0, stream>>>(
        out_proj_w, WHp, WLp, (long long)HID * INTER);
    wmma_gemm_kernel<0, 4><<<dim3(HID / 64, ML / 64), blk, 0, stream>>>(
        AHp, ALp, WHp, WLp, out, nullptr, ML, INTER, INTER, INTER, HID, 0);
}
